// MMDiTAttention_1451698946430
// MI455X (gfx1250) — matrix operand order repaired and checked by reading, not compiled, not run on hardware
//
#include <hip/hip_runtime.h>
#include <hip/hip_bf16.h>

// ---------------------------------------------------------------------------
// MMDiT joint attention for MI455X (gfx1250, wave32, WMMA 16x16x32 f16).
// Compute-bound (~250 GFLOP vs ~100MB unique traffic) -> all GEMMs via v_wmma.
// This revision adds the CDNA5 Tensor Data Mover to the GEMM: the 64-row W
// tile (shared by all 4 waves of a block) is staged into LDS by TDM
// (tensor_load_to_lds, TENSORcnt), double-buffered so the DMA of k-chunk c+1
// overlaps the WMMAs on chunk c. B-fragments then come from LDS (ds_load_b128)
// removing the 4x redundant global W traffic and the loadcnt-0 stall before
// every WMMA.
// ---------------------------------------------------------------------------

typedef __attribute__((ext_vector_type(16))) _Float16 v16h;
typedef __attribute__((ext_vector_type(8)))  _Float16 v8h;
typedef __attribute__((ext_vector_type(8)))  float    v8f;
typedef __attribute__((ext_vector_type(4)))  unsigned int u32x4;
typedef __attribute__((ext_vector_type(8)))  int      i32x8;
typedef __attribute__((ext_vector_type(4)))  int      i32x4;

#define NTOK 4608   // S_CTX + S_IN
#define DIMD 1024

__device__ __forceinline__ v8f wmma16(v16h a, v16h b, v8f c) {
  // D = A(16x32 f16) x B(32x16 f16) + C(16x16 f32)
  return __builtin_amdgcn_wmma_f32_16x16x32_f16(
      false, a, false, b, (short)0, c, false, false);
}
__device__ __forceinline__ v16h cat8(v8h lo, v8h hi) {
  return __builtin_shufflevector(lo, hi, 0,1,2,3,4,5,6,7,8,9,10,11,12,13,14,15);
}
// A fragment: p = rowptr + kk + aoff, 16 contiguous halves
__device__ __forceinline__ v16h ldA(const _Float16* p) {
  v8h lo = *(const v8h*)p;
  v8h hi = *(const v8h*)(p + 8);
  return cat8(lo, hi);
}
// B fragment: p = rowptr + kk + boff, 16 contiguous halves
__device__ __forceinline__ v16h ldB(const _Float16* p) {
  v8h lo = *(const v8h*)p;
  v8h hi = *(const v8h*)(p + 8);
  return cat8(lo, hi);
}

// ---------------------------------------------------------------------------
// TDM: 2D tile load global->LDS.  rows x cols (f16 elems), row stride = K.
// D# layout per CDNA5 ISA ch.8 (group0 128b, group1 256b, groups 2/3 zero).
// ---------------------------------------------------------------------------
__device__ __forceinline__ void tdm_load_tile_f16(
    const _Float16* gptr, unsigned ldsOff,
    unsigned cols /*tile_dim0*/, unsigned rows /*tile_dim1*/,
    unsigned long long strideElems /*tensor_dim0_stride*/)
{
  const unsigned long long ga = (unsigned long long)(size_t)gptr;
  u32x4 g0;
  g0.x = 1u;                                        // count=1 (valid user D#)
  g0.y = ldsOff;                                    // lds_addr (bytes)
  g0.z = (unsigned)(ga & 0xffffffffu);              // global_addr[31:0]
  g0.w = (unsigned)((ga >> 32) & 0x01ffffffu)       // global_addr[56:32]
       | (2u << 30);                                // type = 2 ("image")
  i32x8 g1;
  g1[0] = 1 << 16;                                  // data_size = 1 -> 2 bytes
  g1[1] = (int)((cols & 0xffffu) << 16);            // tensor_dim0[15:0]
  g1[2] = (int)((cols >> 16) | ((rows & 0xffffu) << 16));  // dim0 hi | dim1 lo
  g1[3] = (int)((rows >> 16) | ((cols & 0xffffu) << 16));  // dim1 hi | tile_dim0
  g1[4] = (int)(rows & 0xffffu);                    // tile_dim1 (tile_dim2 = 0)
  g1[5] = (int)(strideElems & 0xffffffffu);         // tensor_dim0_stride lo32
  g1[6] = (int)((strideElems >> 32) & 0xffffu);     // stride hi16 (dim1_stride=0)
  g1[7] = 0;
  const i32x4 gz = {0, 0, 0, 0};
#if defined(__clang_major__) && __clang_major__ >= 23
  const i32x8 gz8 = {0, 0, 0, 0, 0, 0, 0, 0};
  __builtin_amdgcn_tensor_load_to_lds(g0, g1, gz, gz, gz8, 0);
#else
  __builtin_amdgcn_tensor_load_to_lds(g0, g1, gz, gz, 0);
#endif
}

// ---------------------------------------------------------------------------
// Generic row-major WMMA GEMM: C[M,Nout] = A[M,K] @ W[Nout,K]^T + bias
// Block = 128 threads (4 waves), tile = 64x64, wave = 16 rows x 64 cols.
// W tile staged in LDS by TDM in 64x128-half chunks, double buffered;
// K must be a multiple of 128 (1024 and 4608 both are).
// ---------------------------------------------------------------------------
__global__ void __launch_bounds__(128) gemm_wmma(
    const _Float16* __restrict__ A, const _Float16* __restrict__ W,
    const float* __restrict__ bias,
    float* __restrict__ C, _Float16* __restrict__ C16,
    int M, int Nout, int K)
{
  __shared__ _Float16 ldsW[2][64 * 128];

  const int lane = threadIdx.x & 31;
  const int wave = threadIdx.x >> 5;
  const int m0   = blockIdx.x * 64 + wave * 16;
  const int n0   = blockIdx.y * 64;
  const int aoff = (lane < 16) ? 0 : 16;
  const int boff = (lane < 16) ? 0 : 16;

  const _Float16* Arow  = A + (size_t)(m0 + (lane & 15)) * K;
  const _Float16* Wtile = W + (size_t)n0 * K;

  const int kchunks = K >> 7;   // 128 halves per chunk

  // Prologue: wave 0 DMAs chunk 0 into buffer 0, then publishes via barrier.
  if (wave == 0) {
    tdm_load_tile_f16(Wtile, (unsigned)(size_t)&ldsW[0][0], 128u, 64u,
                      (unsigned long long)K);
    __builtin_amdgcn_s_wait_tensorcnt(0);
  }
  __syncthreads();

  v8f acc[4] = {};
  for (int kc = 0; kc < kchunks; ++kc) {
    const int buf = kc & 1;
    // Kick off DMA of next chunk into the other buffer (overlaps WMMAs below).
    if (wave == 0 && kc + 1 < kchunks)
      tdm_load_tile_f16(Wtile + (size_t)(kc + 1) * 128,
                        (unsigned)(size_t)&ldsW[buf ^ 1][0], 128u, 64u,
                        (unsigned long long)K);

    const int kb = kc << 7;
    __builtin_prefetch(Arow + kb + 256, 0, 1);      // global_prefetch_b8
    v16h a0 = ldA(Arow + kb +  0 + aoff);
    v16h a1 = ldA(Arow + kb + 32 + aoff);
    v16h a2 = ldA(Arow + kb + 64 + aoff);
    v16h a3 = ldA(Arow + kb + 96 + aoff);

    const _Float16* Wl = &ldsW[buf][0] + (lane & 15) * 128;
#pragma unroll
    for (int j = 0; j < 4; ++j) {
      const _Float16* Wr = Wl + j * 16 * 128;
      acc[j] = wmma16(a0, ldB(Wr +  0 + boff), acc[j]);
      acc[j] = wmma16(a1, ldB(Wr + 32 + boff), acc[j]);
      acc[j] = wmma16(a2, ldB(Wr + 64 + boff), acc[j]);
      acc[j] = wmma16(a3, ldB(Wr + 96 + boff), acc[j]);
    }

    if (wave == 0 && kc + 1 < kchunks)
      __builtin_amdgcn_s_wait_tensorcnt(0);         // DMA of next chunk done
    __syncthreads();                                 // publish / protect buffers
  }

  const int col = n0 + (lane & 15);
  const int rb  = m0 + ((lane < 16) ? 0 : 8);
#pragma unroll
  for (int j = 0; j < 4; ++j) {
    const float bv = bias ? bias[col + j * 16] : 0.0f;
#pragma unroll
    for (int r = 0; r < 8; ++r) {
      const float v = acc[j][r] + bv;
      const size_t idx = (size_t)(rb + r) * Nout + col + j * 16;
      if (C)   C[idx]   = v;
      if (C16) C16[idx] = (_Float16)v;
    }
  }
}

// ---------------------------------------------------------------------------
// f32 -> f16 convert (grid-stride)
// ---------------------------------------------------------------------------
__global__ void f32_to_f16(const float* __restrict__ src,
                           _Float16* __restrict__ dst, size_t n)
{
  size_t i = (size_t)blockIdx.x * blockDim.x + threadIdx.x;
  const size_t stride = (size_t)gridDim.x * blockDim.x;
  for (; i < n; i += stride) dst[i] = (_Float16)src[i];
}

// ---------------------------------------------------------------------------
// Fused RMS-norm (over full 1024) + per-head RoPE (head_dim 64, rotate-half).
// One block (256 thr) per token. Writes Q,K f16 [B,NTOK,1024] and
// V transposed f16 [B,1024,NTOK] (so P@V reuses the GEMM as P @ (V^T)^T).
// ---------------------------------------------------------------------------
__global__ void __launch_bounds__(256) rms_rope(
    const float* __restrict__ qkvi, const float* __restrict__ qkvc,
    const float* __restrict__ qs_in, const float* __restrict__ ks_in,
    const float* __restrict__ qs_ctx, const float* __restrict__ ks_ctx,
    _Float16* __restrict__ Q, _Float16* __restrict__ K,
    _Float16* __restrict__ Vt)
{
  const int t = blockIdx.x;
  const int b = t / NTOK;
  const int n = t % NTOK;

  const float* src; const float *qs, *ks;
  if (n < 512) { src = qkvc + ((size_t)b * 512  + n)        * 3072; qs = qs_ctx; ks = ks_ctx; }
  else         { src = qkvi + ((size_t)b * 4096 + (n - 512)) * 3072; qs = qs_in;  ks = ks_in;  }

  __shared__ float sv[1024];
  __shared__ float sred[8];
  __shared__ float srrms;
  const int tid = threadIdx.x;
  const float pos = (float)n;

  for (int pass = 0; pass < 2; ++pass) {
    const float* x  = src + pass * 1024;
    const float* sc = pass ? ks : qs;
    _Float16* dst   = pass ? K  : Q;

    float ss = 0.0f;
#pragma unroll
    for (int it = 0; it < 4; ++it) {
      const int i = tid + it * 256;
      const float v = x[i];
      sv[i] = v;
      ss += v * v;
    }
#pragma unroll
    for (int msk = 16; msk >= 1; msk >>= 1) ss += __shfl_xor(ss, msk);
    if ((tid & 31) == 0) sred[tid >> 5] = ss;
    __syncthreads();
    if (tid == 0) {
      float tot = 0.0f;
#pragma unroll
      for (int w = 0; w < 8; ++w) tot += sred[w];
      srrms = rsqrtf(tot * (1.0f / 1024.0f) + 1e-6f);
    }
    __syncthreads();
    const float rr = srrms;

#pragma unroll
    for (int it = 0; it < 4; ++it) {
      const int i = tid + it * 256;
      const int h = i >> 6, d = i & 63, j = d & 31;
      const int i1 = h * 64 + j, i2 = i1 + 32;
      const float x1 = sv[i1] * rr * sc[i1];
      const float x2 = sv[i2] * rr * sc[i2];
      // inv_freq[j] = 10000^(-j/32) ; ln(10000)/32 = 0.28782313662
      const float ang = pos * __expf(-(float)j * 0.2878231366242557f);
      const float cs = __cosf(ang), sn = __sinf(ang);
      const float o = (d < 32) ? (x1 * cs - x2 * sn) : (x2 * cs + x1 * sn);
      dst[((size_t)b * NTOK + n) * 1024 + i] = (_Float16)o;
    }
    __syncthreads();
  }

  // V -> transposed f16 [B, 1024, NTOK]
#pragma unroll
  for (int it = 0; it < 4; ++it) {
    const int i = tid + it * 256;
    Vt[((size_t)b * 1024 + i) * NTOK + n] = (_Float16)src[2048 + i];
  }
}

// ---------------------------------------------------------------------------
// Attention pass 1: per-row softmax stats (max m, sum l) over all 4608 keys.
// Block = 128 thr (4 waves x 16 query rows). Scores via WMMA, full 1024-dim
// contraction (reference contracts over full DIM, not per-head).
// C-layout: VGPR r, lanes 0-15 -> row r, lanes 16-31 -> row r+8; col = lane&15.
// Row reductions = xor-shuffles over masks 1,2,4,8 (stay within 16-lane half).
// ---------------------------------------------------------------------------
__global__ void __launch_bounds__(128) attn_stats(
    const _Float16* __restrict__ Q, const _Float16* __restrict__ Kmat,
    float* __restrict__ Ms, float* __restrict__ Ls)
{
  const int lane = threadIdx.x & 31;
  const int wave = threadIdx.x >> 5;
  const int b  = blockIdx.y;
  const int q0 = blockIdx.x * 64;
  const int aoff = (lane < 16) ? 0 : 16;
  const int boff = (lane < 16) ? 0 : 16;

  const _Float16* Qrow = Q + ((size_t)b * NTOK + q0 + wave * 16 + (lane & 15)) * 1024;
  const _Float16* Kb   = Kmat + (size_t)b * NTOK * 1024;

  float mrun[8], lrun[8];
#pragma unroll
  for (int r = 0; r < 8; ++r) { mrun[r] = -3.0e38f; lrun[r] = 0.0f; }

  for (int kt = 0; kt < NTOK / 64; ++kt) {
    v8f s[4] = {};
    const _Float16* Krow0 = Kb + (size_t)(kt * 64 + (lane & 15)) * 1024;
    for (int kk = 0; kk < 1024; kk += 32) {
      v16h a = ldA(Qrow + kk + aoff);
#pragma unroll
      for (int j = 0; j < 4; ++j)
        s[j] = wmma16(a, ldB(Krow0 + (size_t)j * 16 * 1024 + kk + boff), s[j]);
    }
#pragma unroll
    for (int r = 0; r < 8; ++r) {
      float t = -3.0e38f;
#pragma unroll
      for (int j = 0; j < 4; ++j) { s[j][r] *= 0.125f; t = fmaxf(t, s[j][r]); }
      t = fmaxf(t, __shfl_xor(t, 1));
      t = fmaxf(t, __shfl_xor(t, 2));
      t = fmaxf(t, __shfl_xor(t, 4));
      t = fmaxf(t, __shfl_xor(t, 8));
      const float mn = fmaxf(mrun[r], t);
      const float alpha = __expf(mrun[r] - mn);
      float rs = 0.0f;
#pragma unroll
      for (int j = 0; j < 4; ++j) rs += __expf(s[j][r] - mn);
      rs += __shfl_xor(rs, 1);
      rs += __shfl_xor(rs, 2);
      rs += __shfl_xor(rs, 4);
      rs += __shfl_xor(rs, 8);
      lrun[r] = lrun[r] * alpha + rs;
      mrun[r] = mn;
    }
  }

  if ((lane & 15) == 0) {
    const int rb = q0 + wave * 16 + ((lane < 16) ? 0 : 8);
#pragma unroll
    for (int r = 0; r < 8; ++r) {
      Ms[(size_t)b * NTOK + rb + r] = mrun[r];
      Ls[(size_t)b * NTOK + rb + r] = lrun[r];
    }
  }
}

// ---------------------------------------------------------------------------
// Attention pass 2: materialize P[b][n][m] = exp(s*scale - m)/l as f16.
// One block per 64x64 score tile; recomputes the tile with WMMA.
// ---------------------------------------------------------------------------
__global__ void __launch_bounds__(128) attn_probs(
    const _Float16* __restrict__ Q, const _Float16* __restrict__ Kmat,
    const float* __restrict__ Ms, const float* __restrict__ Ls,
    _Float16* __restrict__ P)
{
  const int lane = threadIdx.x & 31;
  const int wave = threadIdx.x >> 5;
  const int b  = blockIdx.z;
  const int q0 = blockIdx.x * 64;
  const int k0 = blockIdx.y * 64;
  const int aoff = (lane < 16) ? 0 : 16;
  const int boff = (lane < 16) ? 0 : 16;

  const _Float16* Qrow  = Q    + ((size_t)b * NTOK + q0 + wave * 16 + (lane & 15)) * 1024;
  const _Float16* Krow0 = Kmat + ((size_t)b * NTOK + k0 + (lane & 15)) * 1024;

  v8f s[4] = {};
  for (int kk = 0; kk < 1024; kk += 32) {
    v16h a = ldA(Qrow + kk + aoff);
#pragma unroll
    for (int j = 0; j < 4; ++j)
      s[j] = wmma16(a, ldB(Krow0 + (size_t)j * 16 * 1024 + kk + boff), s[j]);
  }

  const int rb = q0 + wave * 16 + ((lane < 16) ? 0 : 8);
#pragma unroll
  for (int r = 0; r < 8; ++r) {
    const float m  = Ms[(size_t)b * NTOK + rb + r];
    const float il = 1.0f / Ls[(size_t)b * NTOK + rb + r];
#pragma unroll
    for (int j = 0; j < 4; ++j) {
      const float pv = __expf(s[j][r] * 0.125f - m) * il;
      P[((size_t)b * NTOK + rb + r) * NTOK + k0 + j * 16 + (lane & 15)] = (_Float16)pv;
    }
  }
}

// ---------------------------------------------------------------------------
// Host-side orchestration
// ---------------------------------------------------------------------------
extern "C" void kernel_launch(void* const* d_in, const int* in_sizes, int n_in,
                              void* d_out, int out_size, void* d_ws, size_t ws_size,
                              hipStream_t stream) {
  (void)in_sizes; (void)n_in; (void)out_size; (void)ws_size;

  const float* input      = (const float*)d_in[0];
  const float* context    = (const float*)d_in[1];
  const float* W_qkv_in   = (const float*)d_in[2];
  const float* b_qkv_in   = (const float*)d_in[3];
  const float* W_qkv_ctx  = (const float*)d_in[4];
  const float* b_qkv_ctx  = (const float*)d_in[5];
  const float* q_scale_in  = (const float*)d_in[6];
  const float* k_scale_in  = (const float*)d_in[7];
  const float* q_scale_ctx = (const float*)d_in[8];
  const float* k_scale_ctx = (const float*)d_in[9];
  const float* W_out_in   = (const float*)d_in[10];
  const float* b_out_in   = (const float*)d_in[11];
  const float* W_out_ctx  = (const float*)d_in[12];
  const float* b_out_ctx  = (const float*)d_in[13];
  float* out = (float*)d_out;

  const size_t B_ = 2, SIN = 4096, SCTX = 512, D = 1024, N3 = 3072, NT = NTOK;

  char* ws = (char*)d_ws;
  size_t off = 0;
  auto take = [&](size_t bytes) -> void* {
    void* p = ws + off;
    off = (off + bytes + 255) & ~(size_t)255;
    return p;
  };

  _Float16* xf16  = (_Float16*)take(B_ * SIN * D * 2);
  _Float16* cf16  = (_Float16*)take(B_ * SCTX * D * 2);
  _Float16* wqi16 = (_Float16*)take(N3 * D * 2);
  _Float16* wqc16 = (_Float16*)take(N3 * D * 2);
  _Float16* woi16 = (_Float16*)take(D * D * 2);
  _Float16* woc16 = (_Float16*)take(SCTX * D * 2);
  float*    qkvi  = (float*)take(B_ * SIN * N3 * 4);
  float*    qkvc  = (float*)take(B_ * SCTX * N3 * 4);
  _Float16* Qh    = (_Float16*)take(B_ * NT * D * 2);
  _Float16* Kh    = (_Float16*)take(B_ * NT * D * 2);
  _Float16* Vt    = (_Float16*)take(B_ * D * NT * 2);
  float*    Ms    = (float*)take(B_ * NT * 4);
  float*    Ls    = (float*)take(B_ * NT * 4);
  _Float16* Pb    = (_Float16*)take(B_ * NT * NT * 2);
  _Float16* O16   = (_Float16*)take(B_ * NT * D * 2);

  // 1) converts
  f32_to_f16<<<2048, 256, 0, stream>>>(input,     xf16,  B_ * SIN * D);
  f32_to_f16<<<1024, 256, 0, stream>>>(context,   cf16,  B_ * SCTX * D);
  f32_to_f16<<<2048, 256, 0, stream>>>(W_qkv_in,  wqi16, N3 * D);
  f32_to_f16<<<2048, 256, 0, stream>>>(W_qkv_ctx, wqc16, N3 * D);
  f32_to_f16<<<1024, 256, 0, stream>>>(W_out_in,  woi16, D * D);
  f32_to_f16<<<512,  256, 0, stream>>>(W_out_ctx, woc16, SCTX * D);

  // 2) QKV projections
  gemm_wmma<<<dim3((B_ * SIN) / 64, N3 / 64), 128, 0, stream>>>(
      xf16, wqi16, b_qkv_in, qkvi, nullptr, (int)(B_ * SIN), (int)N3, (int)D);
  gemm_wmma<<<dim3((B_ * SCTX) / 64, N3 / 64), 128, 0, stream>>>(
      cf16, wqc16, b_qkv_ctx, qkvc, nullptr, (int)(B_ * SCTX), (int)N3, (int)D);

  // 3) RMS-norm + RoPE -> Q, K, V^T (f16)
  rms_rope<<<(unsigned)(B_ * NT), 256, 0, stream>>>(
      qkvi, qkvc, q_scale_in, k_scale_in, q_scale_ctx, k_scale_ctx, Qh, Kh, Vt);

  // 4) softmax stats
  attn_stats<<<dim3(NT / 64, B_), 128, 0, stream>>>(Qh, Kh, Ms, Ls);

  // 5) P matrix (f16)
  attn_probs<<<dim3(NT / 64, NT / 64, B_), 128, 0, stream>>>(Qh, Kh, Ms, Ls, Pb);

  // 6) O = P @ V  (per batch; W = V^T rows are key-contiguous; K = 4608 = 36 TDM chunks)
  for (int b = 0; b < (int)B_; ++b)
    gemm_wmma<<<dim3(NT / 64, D / 64), 128, 0, stream>>>(
        Pb + (size_t)b * NT * NT, Vt + (size_t)b * D * NT, nullptr,
        nullptr, O16 + (size_t)b * NT * D, (int)NT, (int)D, (int)NT);

  // 7) output projections
  for (int b = 0; b < (int)B_; ++b)
    gemm_wmma<<<dim3(SIN / 64, D / 64), 128, 0, stream>>>(
        O16 + ((size_t)b * NT + SCTX) * D, woi16, b_out_in,
        out + (size_t)b * SIN * D, nullptr, (int)SIN, (int)D, (int)D);

  float* out_ctx = out + B_ * SIN * D;
  for (int b = 0; b < (int)B_; ++b)
    gemm_wmma<<<dim3(SCTX / 64, SCTX / 64), 128, 0, stream>>>(
        O16 + (size_t)b * NT * D, woc16, b_out_ctx,
        out_ctx + (size_t)b * SCTX * SCTX, nullptr, (int)SCTX, (int)SCTX, (int)D);
}
